// SRGAAttention_26577257627905
// MI455X (gfx1250) — matrix operand order repaired and checked by reading, not compiled, not run on hardware
//
#include <hip/hip_runtime.h>
#include <hip/hip_bf16.h>

// ---------------------------------------------------------------------------
// CDNA5 / gfx1250 WMMA attention pipeline (wave32, v_wmma_f32_16x16x32_bf16)
// ---------------------------------------------------------------------------

typedef __attribute__((ext_vector_type(16))) __bf16 v16bf;
typedef __attribute__((ext_vector_type(8)))  float  v8f;

union Frag {            // 16 bf16 = 32 bytes = one WMMA A/B operand (8 VGPRs)
    v16bf v;
    uint4 u[2];
};

__device__ __forceinline__ v8f wmma_bf16(v16bf a, v16bf b, v8f c) {
    return __builtin_amdgcn_wmma_f32_16x16x32_bf16(
        /*neg_a=*/false, a, /*neg_b=*/false, b,
        /*c_mod=*/(short)0, c, /*reuse_a=*/false, /*reuse_b=*/false);
}

// Problem constants
#define BATCH   8
#define SEQ     1024
#define DMODEL  768
#define NHEADS  12
#define HEADD   64
#define QKVCOLS 2304
#define MTOT    (BATCH * SEQ)       // 8192

// ---------------------------------------------------------------------------
// Kernel 0a: convert x (f32) -> bf16, same layout [M][D]
// ---------------------------------------------------------------------------
__global__ void k_convert_x(const float* __restrict__ x, __bf16* __restrict__ xb, int n) {
    for (int i = blockIdx.x * blockDim.x + threadIdx.x; i < n; i += gridDim.x * blockDim.x)
        xb[i] = (__bf16)x[i];
}

// ---------------------------------------------------------------------------
// Kernel 0b: transpose+convert W [K][N] f32 -> WT [N][K] bf16
// ---------------------------------------------------------------------------
__global__ void k_transpose_w(const float* __restrict__ W, __bf16* __restrict__ WT,
                              int K, int N) {
    int total = K * N;
    for (int i = blockIdx.x * blockDim.x + threadIdx.x; i < total; i += gridDim.x * blockDim.x) {
        int k = i / N;
        int n = i - k * N;
        WT[(size_t)n * K + k] = (__bf16)W[i];
    }
}

// ---------------------------------------------------------------------------
// Kernel 1: spectral bias. One block (256 thr) per output pixel.
// bias[b][h*32+w] = sum_{d,i,j} x[b][(h+i-1)*32+(w+j-1)][d] * Wb[d][i][j] + bb
// ---------------------------------------------------------------------------
__global__ __launch_bounds__(256) void k_bias_conv(const float* __restrict__ x,
                                                   const float* __restrict__ Wb,
                                                   const float* __restrict__ bb,
                                                   float* __restrict__ bias) {
    int pix = blockIdx.x;                 // 0..8191
    int b   = pix >> 10;
    int hw  = pix & 1023;
    int h   = hw >> 5;
    int w   = hw & 31;
    float sum = 0.f;
    for (int d = threadIdx.x; d < DMODEL; d += 256) {
        #pragma unroll
        for (int i = 0; i < 3; ++i) {
            #pragma unroll
            for (int j = 0; j < 3; ++j) {
                int hh = h + i - 1, ww = w + j - 1;
                if (hh >= 0 && hh < 32 && ww >= 0 && ww < 32) {
                    sum += x[((size_t)b * SEQ + hh * 32 + ww) * DMODEL + d] *
                           Wb[d * 9 + i * 3 + j];
                }
            }
        }
    }
    #pragma unroll
    for (int off = 16; off > 0; off >>= 1) sum += __shfl_xor(sum, off, 32);
    __shared__ float red[8];
    if ((threadIdx.x & 31) == 0) red[threadIdx.x >> 5] = sum;
    __syncthreads();
    if (threadIdx.x == 0) {
        float t = 0.f;
        #pragma unroll
        for (int i = 0; i < 8; ++i) t += red[i];
        bias[pix] = t + bb[0];
    }
}

// ---------------------------------------------------------------------------
// Kernel 2: QKV GEMM. C[M=8192][N=2304] = xb @ WqkvT^T + bqkv, scattered into
//   Q [bh][n][hd], K [bh][n][hd], V transposed [bh][hd][n]   (all bf16)
// Block = 128x128 tile, 8 waves in 2x4; wave = 64x32 = 4x2 WMMA tiles.
// ---------------------------------------------------------------------------
__global__ __launch_bounds__(256) void k_qkv_gemm(const __bf16* __restrict__ xb,
                                                  const __bf16* __restrict__ WT,
                                                  const float*  __restrict__ bqkv,
                                                  __bf16* __restrict__ Qws,
                                                  __bf16* __restrict__ Kws,
                                                  __bf16* __restrict__ Vt) {
    const int m0    = blockIdx.x * 128;
    const int n0    = blockIdx.y * 128;
    const int wave  = threadIdx.x >> 5;
    const int lane  = threadIdx.x & 31;
    const int wm    = wave >> 2;          // 0..1
    const int wn    = wave & 3;           // 0..3
    const int mbase = m0 + wm * 64;
    const int nbase = n0 + wn * 32;
    const int lrow  = lane & 15;
    const int ahalf = (lane < 16) ? 0 : 8;    // A-fragment K sub-offset / D row+8
    const int bhalf = (lane < 16) ? 0 : 8;    // B-fragment K sub-offset

    const v8f vzero = {0.f, 0.f, 0.f, 0.f, 0.f, 0.f, 0.f, 0.f};
    v8f acc[4][2];
    #pragma unroll
    for (int mt = 0; mt < 4; ++mt)
        #pragma unroll
        for (int nt = 0; nt < 2; ++nt) acc[mt][nt] = vzero;

    for (int kc = 0; kc < DMODEL; kc += 32) {
        Frag a[4], bfr[2];
        #pragma unroll
        for (int mt = 0; mt < 4; ++mt) {
            const __bf16* ap = xb + (size_t)(mbase + mt * 16 + lrow) * DMODEL + kc + ahalf;
            a[mt].u[0] = *reinterpret_cast<const uint4*>(ap);
            a[mt].u[1] = *reinterpret_cast<const uint4*>(ap + 16);
        }
        #pragma unroll
        for (int nt = 0; nt < 2; ++nt) {
            const __bf16* bp = WT + (size_t)(nbase + nt * 16 + lrow) * DMODEL + kc + bhalf;
            bfr[nt].u[0] = *reinterpret_cast<const uint4*>(bp);
            bfr[nt].u[1] = *reinterpret_cast<const uint4*>(bp + 16);
        }
        #pragma unroll
        for (int mt = 0; mt < 4; ++mt)
            #pragma unroll
            for (int nt = 0; nt < 2; ++nt)
                acc[mt][nt] = wmma_bf16(a[mt].v, bfr[nt].v, acc[mt][nt]);
    }

    // Epilogue: add bias, scatter to Q / K / V^T as bf16
    #pragma unroll
    for (int mt = 0; mt < 4; ++mt) {
        #pragma unroll
        for (int nt = 0; nt < 2; ++nt) {
            const int col = nbase + nt * 16 + lrow;
            const float bias = bqkv[col];
            #pragma unroll
            for (int r = 0; r < 8; ++r) {
                const int row = mbase + mt * 16 + r + ahalf;   // token index
                const int b   = row >> 10;
                const int n   = row & 1023;
                const float v = acc[mt][nt][r] + bias;
                const __bf16 vb = (__bf16)v;
                if (col < DMODEL) {                            // Q
                    int hh = col >> 6, o = col & 63;
                    Qws[(((size_t)b * NHEADS + hh) * SEQ + n) * HEADD + o] = vb;
                } else if (col < 2 * DMODEL) {                 // K
                    int jo = col - DMODEL;
                    int hh = jo >> 6, o = jo & 63;
                    Kws[(((size_t)b * NHEADS + hh) * SEQ + n) * HEADD + o] = vb;
                } else {                                       // V (transposed)
                    int jo = col - 2 * DMODEL;
                    int hh = jo >> 6, o = jo & 63;
                    Vt[(((size_t)b * NHEADS + hh) * HEADD + o) * SEQ + n] = vb;
                }
            }
        }
    }
}

// ---------------------------------------------------------------------------
// Kernel 3: flash attention. grid = (B*nh, SEQ/128); each wave owns 16 queries.
// Streams 32-key tiles: S = Q K^T (4 WMMA), online softmax, P V (4 WMMA).
// P goes D-layout -> A-layout through per-wave LDS (same-wave DS is in-order).
// ---------------------------------------------------------------------------
__global__ __launch_bounds__(256) void k_attention(const __bf16* __restrict__ Qws,
                                                   const __bf16* __restrict__ Kws,
                                                   const __bf16* __restrict__ Vt,
                                                   const float*  __restrict__ biasArr,
                                                   __bf16* __restrict__ attn_out) {
    const int bh   = blockIdx.x;            // 0..95
    const int q0   = blockIdx.y * 128;
    const int b    = bh / NHEADS;
    const int h    = bh % NHEADS;
    const int wave = threadIdx.x >> 5;
    const int lane = threadIdx.x & 31;
    const int lrow = lane & 15;
    const int ahalf = (lane < 16) ? 0 : 8;
    const int bhalf = (lane < 16) ? 0 : 8;
    const float scale = 0.125f;             // hd^-0.5 = 1/8

    __shared__ __bf16 pstage[8][16][32];    // per-wave P transpose staging (8 KB)

    // Q fragments for this wave's 16 queries (two hd chunks of 32)
    const __bf16* Qbase = Qws + ((size_t)bh * SEQ + q0 + wave * 16) * HEADD;
    Frag qf[2];
    #pragma unroll
    for (int c = 0; c < 2; ++c) {
        const __bf16* qp = Qbase + (size_t)lrow * HEADD + c * 32 + ahalf;
        qf[c].u[0] = *reinterpret_cast<const uint4*>(qp);
        qf[c].u[1] = *reinterpret_cast<const uint4*>(qp + 16);
    }

    const v8f vzero = {0.f, 0.f, 0.f, 0.f, 0.f, 0.f, 0.f, 0.f};
    v8f O[4];
    #pragma unroll
    for (int j = 0; j < 4; ++j) O[j] = vzero;
    float mrun[8], lrun[8];
    #pragma unroll
    for (int r = 0; r < 8; ++r) { mrun[r] = -3.0e38f; lrun[r] = 0.f; }

    const __bf16* Kbh  = Kws + (size_t)bh * SEQ * HEADD;
    const __bf16* Vbh  = Vt  + (size_t)bh * HEADD * SEQ;
    const float*  bs   = biasArr + (size_t)b * SEQ;

    for (int k0 = 0; k0 < SEQ; k0 += 32) {
        // ---- scores: two 16-key sub-tiles, accumulate over hd chunks
        v8f S[2];
        #pragma unroll
        for (int s = 0; s < 2; ++s) {
            v8f t = vzero;
            #pragma unroll
            for (int c = 0; c < 2; ++c) {
                Frag kf;
                const __bf16* kp = Kbh + (size_t)(k0 + s * 16 + lrow) * HEADD + c * 32 + bhalf;
                kf.u[0] = *reinterpret_cast<const uint4*>(kp);
                kf.u[1] = *reinterpret_cast<const uint4*>(kp + 16);
                t = wmma_bf16(qf[c].v, kf.v, t);
            }
            S[s] = t;
        }
        const float bias0 = bs[k0 + lrow];
        const float bias1 = bs[k0 + 16 + lrow];

        // ---- online softmax per row (row lives across 16 lanes)
        #pragma unroll
        for (int r = 0; r < 8; ++r) {
            float s0 = S[0][r] * scale + bias0;
            float s1 = S[1][r] * scale + bias1;
            float mx = fmaxf(s0, s1);
            #pragma unroll
            for (int off = 1; off < 16; off <<= 1) mx = fmaxf(mx, __shfl_xor(mx, off, 32));
            const float mnew  = fmaxf(mrun[r], mx);
            const float alpha = __expf(mrun[r] - mnew);
            const float p0 = __expf(s0 - mnew);
            const float p1 = __expf(s1 - mnew);
            float rs = p0 + p1;
            #pragma unroll
            for (int off = 1; off < 16; off <<= 1) rs += __shfl_xor(rs, off, 32);
            lrun[r] = lrun[r] * alpha + rs;
            mrun[r] = mnew;
            #pragma unroll
            for (int j = 0; j < 4; ++j) O[j][r] *= alpha;
            pstage[wave][r + ahalf][lrow]      = (__bf16)p0;
            pstage[wave][r + ahalf][16 + lrow] = (__bf16)p1;
        }

        // ---- reload P in A-fragment layout (same-wave LDS ops are in-order)
        Frag pf;
        const __bf16* pp = &pstage[wave][lrow][0] + ahalf;
        pf.u[0] = *reinterpret_cast<const uint4*>(pp);
        pf.u[1] = *reinterpret_cast<const uint4*>(pp + 16);

        // ---- O += P @ V  (V pre-transposed: each lane loads two 8-key runs of its hd col)
        #pragma unroll
        for (int j = 0; j < 4; ++j) {
            Frag vf;
            const __bf16* vp = Vbh + (size_t)(j * 16 + lrow) * SEQ + k0 + bhalf;
            vf.u[0] = *reinterpret_cast<const uint4*>(vp);
            vf.u[1] = *reinterpret_cast<const uint4*>(vp + 16);
            O[j] = wmma_bf16(pf.v, vf.v, O[j]);
        }
    }

    // ---- normalize and store to [B][N][D] bf16 (D col = h*64 + j*16 + lane%16)
    #pragma unroll
    for (int j = 0; j < 4; ++j) {
        const int d = h * HEADD + j * 16 + lrow;
        #pragma unroll
        for (int r = 0; r < 8; ++r) {
            const int q = q0 + wave * 16 + r + ahalf;
            const float o = O[j][r] / lrun[r];
            attn_out[((size_t)b * SEQ + q) * DMODEL + d] = (__bf16)o;
        }
    }
}

// ---------------------------------------------------------------------------
// Kernel 4: output projection. out[8192][768] f32 = attn_bf @ WprojT^T + bproj
// ---------------------------------------------------------------------------
__global__ __launch_bounds__(256) void k_proj_gemm(const __bf16* __restrict__ A,
                                                   const __bf16* __restrict__ WT,
                                                   const float*  __restrict__ bproj,
                                                   float* __restrict__ out) {
    const int m0    = blockIdx.x * 128;
    const int n0    = blockIdx.y * 128;
    const int wave  = threadIdx.x >> 5;
    const int lane  = threadIdx.x & 31;
    const int wm    = wave >> 2;
    const int wn    = wave & 3;
    const int mbase = m0 + wm * 64;
    const int nbase = n0 + wn * 32;
    const int lrow  = lane & 15;
    const int ahalf = (lane < 16) ? 0 : 8;
    const int bhalf = (lane < 16) ? 0 : 8;

    const v8f vzero = {0.f, 0.f, 0.f, 0.f, 0.f, 0.f, 0.f, 0.f};
    v8f acc[4][2];
    #pragma unroll
    for (int mt = 0; mt < 4; ++mt)
        #pragma unroll
        for (int nt = 0; nt < 2; ++nt) acc[mt][nt] = vzero;

    for (int kc = 0; kc < DMODEL; kc += 32) {
        Frag a[4], bfr[2];
        #pragma unroll
        for (int mt = 0; mt < 4; ++mt) {
            const __bf16* ap = A + (size_t)(mbase + mt * 16 + lrow) * DMODEL + kc + ahalf;
            a[mt].u[0] = *reinterpret_cast<const uint4*>(ap);
            a[mt].u[1] = *reinterpret_cast<const uint4*>(ap + 16);
        }
        #pragma unroll
        for (int nt = 0; nt < 2; ++nt) {
            const __bf16* bp = WT + (size_t)(nbase + nt * 16 + lrow) * DMODEL + kc + bhalf;
            bfr[nt].u[0] = *reinterpret_cast<const uint4*>(bp);
            bfr[nt].u[1] = *reinterpret_cast<const uint4*>(bp + 16);
        }
        #pragma unroll
        for (int mt = 0; mt < 4; ++mt)
            #pragma unroll
            for (int nt = 0; nt < 2; ++nt)
                acc[mt][nt] = wmma_bf16(a[mt].v, bfr[nt].v, acc[mt][nt]);
    }

    #pragma unroll
    for (int mt = 0; mt < 4; ++mt) {
        #pragma unroll
        for (int nt = 0; nt < 2; ++nt) {
            const int col = nbase + nt * 16 + lrow;
            const float bias = bproj[col];
            #pragma unroll
            for (int r = 0; r < 8; ++r) {
                const int row = mbase + mt * 16 + r + ahalf;
                out[(size_t)row * DMODEL + col] = acc[mt][nt][r] + bias;
            }
        }
    }
}

// ---------------------------------------------------------------------------
// Launch
// ---------------------------------------------------------------------------
extern "C" void kernel_launch(void* const* d_in, const int* in_sizes, int n_in,
                              void* d_out, int out_size, void* d_ws, size_t ws_size,
                              hipStream_t stream) {
    const float* x     = (const float*)d_in[0];
    const float* Wqkv  = (const float*)d_in[3];
    const float* bqkv  = (const float*)d_in[4];
    const float* Wproj = (const float*)d_in[5];
    const float* bproj = (const float*)d_in[6];
    const float* Wbias = (const float*)d_in[7];
    const float* bbias = (const float*)d_in[8];
    float* out = (float*)d_out;

    // Workspace partition (all 256B-aligned)
    char* ws = (char*)d_ws;
    size_t off = 0;
    auto alloc = [&](size_t bytes) { char* p = ws + off; off = (off + bytes + 255) & ~(size_t)255; return p; };
    __bf16* xb      = (__bf16*)alloc((size_t)MTOT * DMODEL * 2);          // 12.6 MB
    __bf16* WqkvT   = (__bf16*)alloc((size_t)QKVCOLS * DMODEL * 2);       //  3.5 MB
    __bf16* WprojT  = (__bf16*)alloc((size_t)DMODEL * DMODEL * 2);        //  1.2 MB
    float*  biasArr = (float*) alloc((size_t)BATCH * SEQ * 4);            //  32 KB
    __bf16* Qws     = (__bf16*)alloc((size_t)MTOT * DMODEL * 2);          // 12.6 MB
    __bf16* Kws     = (__bf16*)alloc((size_t)MTOT * DMODEL * 2);          // 12.6 MB
    __bf16* Vt      = (__bf16*)alloc((size_t)MTOT * DMODEL * 2);          // 12.6 MB
    __bf16* attn_bf = (__bf16*)alloc((size_t)MTOT * DMODEL * 2);          // 12.6 MB
    (void)ws_size;

    // 0) conversions / transposes
    k_convert_x<<<4096, 256, 0, stream>>>(x, xb, MTOT * DMODEL);
    k_transpose_w<<<2048, 256, 0, stream>>>(Wqkv, WqkvT, DMODEL, QKVCOLS);
    k_transpose_w<<<1024, 256, 0, stream>>>(Wproj, WprojT, DMODEL, DMODEL);

    // 1) spectral bias map
    k_bias_conv<<<BATCH * SEQ, 256, 0, stream>>>(x, Wbias, bbias, biasArr);

    // 2) QKV GEMM -> Q, K, V^T (bf16)
    k_qkv_gemm<<<dim3(MTOT / 128, QKVCOLS / 128), 256, 0, stream>>>(
        xb, WqkvT, bqkv, Qws, Kws, Vt);

    // 3) flash attention
    k_attention<<<dim3(BATCH * NHEADS, SEQ / 128), 256, 0, stream>>>(
        Qws, Kws, Vt, biasArr, attn_bf);

    // 4) output projection -> f32 d_out
    k_proj_gemm<<<dim3(MTOT / 128, DMODEL / 128), 256, 0, stream>>>(
        attn_bf, WprojT, bproj, out);
}
